// TiedRowAxialAttention_77756087927090
// MI455X (gfx1250) — matrix operand order repaired and checked by reading, not compiled, not run on hardware
//
#include <hip/hip_runtime.h>
#include <hip/hip_bf16.h>

// ---------------------------------------------------------------------------
// TiedRowAxialAttention on MI455X (gfx1250)
//   wave32 + V_WMMA_F32_16X16X32_BF16, async global->LDS double buffering
//   (GLOBAL_LOAD_ASYNC_TO_LDS_B128 / s_wait_asynccnt), fragment-major LDS.
//
// Pipeline:
//   1. cast x -> bf16, cast W0 -> bf16, repack W_qkv rows (d,k,h) -> (k,h,d)
//   2. QKV GEMM [65536x512]@[512x1536] -> scatter Q/K [bh][t][r*64+d], V^T
//   3. Scores per bh: Q[256x4096] @ K^T -> S f32
//   4. Softmax (scale 1/64) -> attn bf16
//   5. AV per bh: attn[256x256] @ V -> O[(b r)][t][(h d)] bf16
//   6. Output GEMM [65536x512]@[512x512] -> d_out f32
// ---------------------------------------------------------------------------

#define BATCH 4
#define ROWS  64
#define TT    256
#define DIM   512
#define HEADS 8
#define DHEAD 64
#define BR    (BATCH * ROWS)      // 256
#define MTOK  (BR * TT)           // 65536
#define RD    (ROWS * DHEAD)      // 4096
#define NBH   (BATCH * HEADS)     // 32

typedef __attribute__((ext_vector_type(16))) __bf16 bf16x16;
typedef __attribute__((ext_vector_type(8)))  float  f32x8;

struct Pack32 { uint4 lo, hi; };

__device__ __forceinline__ __bf16 f2bf(float f) {
    unsigned u = __builtin_bit_cast(unsigned, f);
    u += 0x7FFFu + ((u >> 16) & 1u);
    unsigned short h = (unsigned short)(u >> 16);
    return __builtin_bit_cast(__bf16, h);
}

// Async 16B copy: global -> LDS, tracked by ASYNCcnt.
// vdst = LDS byte offset (low 32 bits of a flat shared pointer), vaddr = 64-bit
// global address.
__device__ __forceinline__ void async_copy_b128(void* lds_ptr, const void* gptr) {
    const unsigned lds_off = (unsigned)(uintptr_t)lds_ptr;
    asm volatile("global_load_async_to_lds_b128 %0, %1, off"
                 :: "v"(lds_off), "v"(gptr) : "memory");
}
__device__ __forceinline__ void wait_async_le4() {
    asm volatile("s_wait_asynccnt 0x4" ::: "memory");
}
__device__ __forceinline__ void wait_async_0() {
    asm volatile("s_wait_asynccnt 0x0" ::: "memory");
}

// ---------------------------------------------------------------------------
// LDS fragment layout (per 32-elem K slab):
//   tile t (16 rows/cols of the block tile), chunk c in {0,1}, lane l:
//   16B chunk stored at element offset ((t*2 + c)*32 + l) * 8.
//   => fragment read: lane l does two ds_load_b128 at t*512 + l*8 (+256),
//      lanes spaced 16B (minimum-phase LDS access).
// A-chunk (16x32 MxK frag):  row = t*16 + (l&15), k = c*16 + (l>>4)*8
// B-chunk (32x16 KxN frag):  col = t*16 + (l&15), k = c*16 + (l>>4)*8
// ---------------------------------------------------------------------------

#define TILE_ELEMS 4096   // 128 rows/cols x 32 K of bf16 = 8KB per buffer

__device__ __forceinline__ void stage_tile(const __bf16* __restrict__ A, int lda, int m0,
                                           const __bf16* __restrict__ Bt, int ldb, int n0,
                                           int k0, __bf16* sa, __bf16* sb, int tid) {
#pragma unroll
    for (int h = 0; h < 2; ++h) {
        const int q = tid + h * 256;          // chunk id, 0..511
        const int t = q >> 6;                 // 16-row tile
        const int c = (q >> 5) & 1;           // chunk within fragment
        const int l = q & 31;                 // lane slot
        const int rl = l & 15, hf = l >> 4;

        const __bf16* ga = A + (m0 + t * 16 + rl) * lda + k0 + c * 16 + hf * 8;
        async_copy_b128(sa + q * 8, ga);

        const __bf16* gb = Bt + (n0 + t * 16 + rl) * ldb + k0 + c * 16 + hf * 8;
        async_copy_b128(sb + q * 8, gb);
    }
}

__device__ __forceinline__ bf16x16 lds_frag(const __bf16* sm, int t, int lane) {
    const __bf16* p = sm + t * 512 + lane * 8;
    Pack32 pk;
    pk.lo = *(const uint4*)(p);
    pk.hi = *(const uint4*)(p + 256);
    return __builtin_bit_cast(bf16x16, pk);
}

// Core GEMM: 256 threads = 8 waves; block tile 128(M) x 128(N); K step 32.
// Waves: 4 in M x 2 in N; each wave owns 32(M) x 64(N) = 2x4 WMMA tiles.
// Double-buffered LDS staging via async copies; per k-step all 12 ds_loads
// are issued before the 8 WMMAs so dscnt waits can be partial.
template <typename Epi>
__device__ __forceinline__ void gemm_core_lds(const __bf16* __restrict__ A, int lda,
                                              const __bf16* __restrict__ Bt, int ldb,
                                              int K, int m0, int n0,
                                              __bf16* smA, __bf16* smB, Epi epi) {
    const int tid  = threadIdx.x;
    const int wave = tid >> 5;
    const int lane = tid & 31;
    const int wm = wave & 3;      // m position (x32 rows)
    const int wn = wave >> 2;     // n position (x64 cols)

    f32x8 acc[2][4] = {};

    const int nk = K >> 5;
    stage_tile(A, lda, m0, Bt, ldb, n0, 0, smA, smB, tid);

    for (int kt = 0; kt < nk; ++kt) {
        if (kt + 1 < nk) {
            stage_tile(A, lda, m0, Bt, ldb, n0, (kt + 1) << 5,
                       smA + ((kt + 1) & 1) * TILE_ELEMS,
                       smB + ((kt + 1) & 1) * TILE_ELEMS, tid);
            wait_async_le4();       // batch kt has landed; batch kt+1 in flight
        } else {
            wait_async_0();
        }
        __syncthreads();            // whole tile kt visible to all waves

        const __bf16* sa = smA + (kt & 1) * TILE_ELEMS;
        const __bf16* sb = smB + (kt & 1) * TILE_ELEMS;

        // issue all fragment loads for this k-step first...
        bf16x16 af[2], bfr[4];
#pragma unroll
        for (int it = 0; it < 2; ++it)
            af[it] = lds_frag(sa, wm * 2 + it, lane);
#pragma unroll
        for (int j = 0; j < 4; ++j)
            bfr[j] = lds_frag(sb, wn * 4 + j, lane);

        // ...then run the 8 WMMAs as one dense block
#pragma unroll
        for (int j = 0; j < 4; ++j)
#pragma unroll
            for (int it = 0; it < 2; ++it)
                acc[it][j] = __builtin_amdgcn_wmma_f32_16x16x32_bf16(
                    false, af[it], false, bfr[j], (short)0, acc[it][j], false, false);

        __syncthreads();            // done reading buf kt; it may be overwritten
    }

    const int nl = lane & 15;
    const int mh = lane >> 4;
#pragma unroll
    for (int it = 0; it < 2; ++it)
#pragma unroll
        for (int j = 0; j < 4; ++j)
#pragma unroll
            for (int r = 0; r < 8; ++r)
                epi(m0 + wm * 32 + it * 16 + r + 8 * mh,
                    n0 + wn * 64 + j * 16 + nl, acc[it][j][r]);
}

// ---------------------------------------------------------------------------
// Prep kernels
// ---------------------------------------------------------------------------

__global__ void k_cast_bf16(const float* __restrict__ in, __bf16* __restrict__ out, int n) {
    int i = blockIdx.x * blockDim.x + threadIdx.x;
    for (; i < n; i += gridDim.x * blockDim.x)
        out[i] = f2bf(in[i]);
}

// W_qkv rows ordered c = d*24 + k*8 + h  ->  packed rows c2 = (k*8+h)*64 + d
__global__ void k_pack_wqkv(const float* __restrict__ W, __bf16* __restrict__ Wp) {
    int idx = blockIdx.x * blockDim.x + threadIdx.x;   // over 1536*512
    const int c2 = idx >> 9;
    const int kk = idx & 511;
    const int kh = c2 >> 6;
    const int d  = c2 & 63;
    Wp[idx] = f2bf(W[(d * 24 + kh) * DIM + kk]);
}

// ---------------------------------------------------------------------------
// GEMM kernels
// ---------------------------------------------------------------------------

__global__ void k_gemm_qkv(const __bf16* __restrict__ xb, const __bf16* __restrict__ Wp,
                           __bf16* __restrict__ Q, __bf16* __restrict__ Kk,
                           __bf16* __restrict__ Vt) {
    __shared__ __align__(16) __bf16 smA[2 * TILE_ELEMS];
    __shared__ __align__(16) __bf16 smB[2 * TILE_ELEMS];
    gemm_core_lds(xb, DIM, Wp, DIM, DIM, blockIdx.y * 128, blockIdx.x * 128, smA, smB,
        [=](int m, int n, float v) {
            const int br = m >> 8, i = m & 255;
            const int b = br >> 6, r = br & 63;
            const int kq = n >> 9;            // 0=q, 1=k, 2=v
            const int hd = n & 511;           // h*64 + d
            const int bh = b * HEADS + (hd >> 6);
            const int rd = (r << 6) + (hd & 63);
            if (kq == 0)      Q [(bh * TT + i) * RD + rd] = f2bf(v);
            else if (kq == 1) Kk[(bh * TT + i) * RD + rd] = f2bf(v);
            else              Vt[(bh * RD + rd) * TT + i] = f2bf(v);
        });
}

__global__ void k_gemm_scores(const __bf16* __restrict__ Q, const __bf16* __restrict__ Kk,
                              float* __restrict__ S) {
    __shared__ __align__(16) __bf16 smA[2 * TILE_ELEMS];
    __shared__ __align__(16) __bf16 smB[2 * TILE_ELEMS];
    const int bh = blockIdx.z;
    const __bf16* A  = Q  + (size_t)bh * TT * RD;
    const __bf16* Bt = Kk + (size_t)bh * TT * RD;
    float* Sb = S + (size_t)bh * TT * TT;
    gemm_core_lds(A, RD, Bt, RD, RD, blockIdx.y * 128, blockIdx.x * 128, smA, smB,
        [=](int m, int n, float v) { Sb[m * TT + n] = v; });
}

__global__ void k_softmax(const float* __restrict__ S, __bf16* __restrict__ attn) {
    __shared__ float red[TT];
    const int row = blockIdx.x;                  // bh*256 + i
    const int tid = threadIdx.x;
    const float scale = 1.0f / 64.0f;            // (rows*dim_head)^-0.5

    float v = S[(size_t)row * TT + tid] * scale;
    red[tid] = v;
    __syncthreads();
#pragma unroll
    for (int s = 128; s > 0; s >>= 1) {
        if (tid < s) red[tid] = fmaxf(red[tid], red[tid + s]);
        __syncthreads();
    }
    const float mx = red[0];
    __syncthreads();

    const float e = __expf(v - mx);
    red[tid] = e;
    __syncthreads();
#pragma unroll
    for (int s = 128; s > 0; s >>= 1) {
        if (tid < s) red[tid] += red[tid + s];
        __syncthreads();
    }
    const float inv = 1.0f / red[0];
    attn[(size_t)row * TT + tid] = f2bf(e * inv);
}

__global__ void k_gemm_av(const __bf16* __restrict__ attn, const __bf16* __restrict__ Vt,
                          __bf16* __restrict__ O2) {
    __shared__ __align__(16) __bf16 smA[2 * TILE_ELEMS];
    __shared__ __align__(16) __bf16 smB[2 * TILE_ELEMS];
    const int bh = blockIdx.z;
    const int b = bh >> 3, h = bh & 7;
    const __bf16* A  = attn + (size_t)bh * TT * TT;   // lda = 256
    const __bf16* Bt = Vt   + (size_t)bh * RD * TT;   // [4096][256]
    gemm_core_lds(A, TT, Bt, TT, TT, blockIdx.y * 128, blockIdx.x * 128, smA, smB,
        [=](int m, int n, float v) {
            const int r = n >> 6, d = n & 63;
            O2[(((b * ROWS + r) * TT + m) << 9) + (h << 6) + d] = f2bf(v);
        });
}

__global__ void k_gemm_out(const __bf16* __restrict__ O2, const __bf16* __restrict__ W0b,
                           float* __restrict__ out) {
    __shared__ __align__(16) __bf16 smA[2 * TILE_ELEMS];
    __shared__ __align__(16) __bf16 smB[2 * TILE_ELEMS];
    gemm_core_lds(O2, DIM, W0b, DIM, DIM, blockIdx.y * 128, blockIdx.x * 128, smA, smB,
        [=](int m, int n, float v) { out[(size_t)m * DIM + n] = v; });
}

// ---------------------------------------------------------------------------

extern "C" void kernel_launch(void* const* d_in, const int* in_sizes, int n_in,
                              void* d_out, int out_size, void* d_ws, size_t ws_size,
                              hipStream_t stream) {
    const float* x     = (const float*)d_in[0];   // [65536, 512]
    const float* W_qkv = (const float*)d_in[1];   // [1536, 512]
    const float* W0    = (const float*)d_in[2];   // [512, 512]
    float* out = (float*)d_out;

    char* w = (char*)d_ws;
    __bf16* xb  = (__bf16*)w;  w += (size_t)MTOK * DIM * 2;     //  64 MB
    __bf16* Q   = (__bf16*)w;  w += (size_t)NBH * TT * RD * 2;  //  64 MB
    __bf16* Kk  = (__bf16*)w;  w += (size_t)NBH * TT * RD * 2;  //  64 MB
    __bf16* Vt  = (__bf16*)w;  w += (size_t)NBH * RD * TT * 2;  //  64 MB
    float*  S   = (float*)w;   w += (size_t)NBH * TT * TT * 4;  //   8 MB
    __bf16* At  = (__bf16*)w;  w += (size_t)NBH * TT * TT * 2;  //   4 MB
    __bf16* Wp  = (__bf16*)w;  w += (size_t)3 * DIM * DIM * 2;  // 1.5 MB
    __bf16* W0b = (__bf16*)w;  w += (size_t)DIM * DIM * 2;      // 0.5 MB
    __bf16* O2  = xb;  // xb dead after QKV GEMM; reuse its slab

    k_cast_bf16<<<dim3(MTOK * DIM / 256), dim3(256), 0, stream>>>(x, xb, MTOK * DIM);
    k_cast_bf16<<<dim3(DIM * DIM / 256), dim3(256), 0, stream>>>(W0, W0b, DIM * DIM);
    k_pack_wqkv<<<dim3(3 * DIM * DIM / 256), dim3(256), 0, stream>>>(W_qkv, Wp);

    // QKV: grid (1536/128, 65536/128)
    k_gemm_qkv<<<dim3(12, 512), dim3(256), 0, stream>>>(xb, Wp, Q, Kk, Vt);

    // scores: per bh, grid (256/128, 256/128, 32)
    k_gemm_scores<<<dim3(2, 2, NBH), dim3(256), 0, stream>>>(Q, Kk, S);

    // softmax: one block per score row
    k_softmax<<<dim3(NBH * TT), dim3(TT), 0, stream>>>(S, At);

    // AV: per bh, grid (4096/128, 256/128, 32)
    k_gemm_av<<<dim3(32, 2, NBH), dim3(256), 0, stream>>>(At, Vt, O2);

    // output projection: grid (512/128, 65536/128)
    k_gemm_out<<<dim3(4, 512), dim3(256), 0, stream>>>(O2, W0b, out);
}
